// HDLPKT_21320217657953
// MI455X (gfx1250) — compile-verified, matrix operand order checked by reading, not run on hardware
//
#include <hip/hip_runtime.h>
#include <hip/hip_bf16.h>

#define DEV __device__ __forceinline__

typedef _Float16 v16h __attribute__((ext_vector_type(16)));
typedef _Float16 v8h  __attribute__((ext_vector_type(8)));
typedef float    v8f  __attribute__((ext_vector_type(8)));

DEV v8f zero8() { v8f z; for (int i = 0; i < 8; ++i) z[i] = 0.f; return z; }
DEV v8f wmma32(v16h a, v16h b, v8f c) {
  return __builtin_amdgcn_wmma_f32_16x16x32_f16(false, a, false, b, (short)0, c, false, false);
}
DEV float sigf(float x) { return 1.f / (1.f + __expf(-x)); }

// ---- RNG (counter hash; reference uses threefry which we cannot reproduce) ----
DEV unsigned pcg(unsigned v) {
  v = v * 747796405u + 2891336453u;
  unsigned w = ((v >> ((v >> 28) + 4u)) ^ v) * 277803737u;
  return (w >> 22) ^ w;
}
DEV float u01(unsigned s) { return (pcg(s) >> 8) * (1.0f / 16777216.0f) + 1e-7f; }
DEV float nrm(unsigned s) {
  float u1 = u01(2u * s + 11u), u2 = u01(2u * s + 77u);
  return sqrtf(-2.f * logf(u1)) * cosf(6.28318530718f * u2);
}

// ---- Fast WMMA 16x32 f16 operand tile loader (A layout; B symmetric).
// Requires: 16-row tile fully in range, 32 k-values in range, base 16B-aligned,
// sr (elements) a multiple of 8. Two 16-byte vector loads per lane.
DEV v16h load_tile(const _Float16* base, long sr, int r0, int k0) {
  int l = threadIdx.x & 31;
  const _Float16* p = base + (long)(r0 + (l & 15)) * sr + k0 + (l >> 4) * 8;
  v8h lo = *(const v8h*)p;
  v8h hi = *(const v8h*)(p + 16);
  v16h out;
#pragma unroll
  for (int i = 0; i < 8; ++i) { out[i] = lo[i]; out[8 + i] = hi[i]; }
  return out;
}

// =====================================================================
// Generic WMMA GEMM, 2-wide N register blocking (M%16==0, N%32==0, Kd%32==0):
// D[m,n] = act( sum_k A[m,k]*W[n,k] + bias[n] )
// Each wave computes a 16x32 strip: one A tile feeds two independent WMMAs.
// =====================================================================
__global__ void gemm_wmma_kernel(const _Float16* __restrict__ A, long a_rs, long a_bs,
                                 const _Float16* __restrict__ W, long w_ns, long w_bs,
                                 const float* __restrict__ bias,
                                 float* __restrict__ C, long c_rs, long c_bs,
                                 _Float16* __restrict__ C16, long c16_rs, long c16_bs,
                                 int M, int N, int Kd, int act) {
  int wave = (blockIdx.x * blockDim.x + threadIdx.x) >> 5;
  int tn2 = N >> 5, tm = M >> 4;
  if (wave >= tm * tn2) return;
  int mt = wave / tn2, ntp = wave % tn2;
  long bz = blockIdx.y;
  const _Float16* Ab = A + bz * a_bs + (long)(mt * 16) * a_rs;
  const _Float16* W0 = W + bz * w_bs + (long)(ntp * 32) * w_ns;
  const _Float16* W1 = W0 + (long)16 * w_ns;
  v8f acc0 = zero8(), acc1 = zero8();
  for (int k0 = 0; k0 < Kd; k0 += 32) {
    if ((k0 & 63) == 0 && k0 + 64 < Kd) {
      __builtin_prefetch(W0 + k0 + 64, 0, 1);
      __builtin_prefetch(W1 + k0 + 64, 0, 1);
    }
    v16h a = load_tile(Ab, a_rs, 0, k0);
    v16h w0 = load_tile(W0, w_ns, 0, k0);
    v16h w1 = load_tile(W1, w_ns, 0, k0);
    acc0 = wmma32(a, w0, acc0);
    acc1 = wmma32(a, w1, acc1);
  }
  int l = threadIdx.x & 31;
  int col0 = ntp * 32 + (l & 15);
  long rbase = mt * 16 + (l >> 4) * 8;
  float bv0 = bias ? bias[col0] : 0.f;
  float bv1 = bias ? bias[col0 + 16] : 0.f;
#pragma unroll
  for (int r = 0; r < 8; ++r) {
    float v0 = acc0[r] + bv0, v1 = acc1[r] + bv1;
    if (act == 1) { v0 = fmaxf(v0, 0.f); v1 = fmaxf(v1, 0.f); }
    else if (act == 2) { v0 = sigf(v0); v1 = sigf(v1); }
    else if (act == 3) { v0 = tanhf(v0); v1 = tanhf(v1); }
    if (C) {
      float* c = C + bz * c_bs + (rbase + r) * c_rs + col0;
      c[0] = v0; c[16] = v1;
    }
    if (C16) {
      _Float16* c = C16 + bz * c16_bs + (rbase + r) * c16_rs + col0;
      c[0] = (_Float16)v0; c[16] = (_Float16)v1;
    }
  }
}

// ---- elementwise helpers --------------------------------------------
__global__ void f32_to_f16_kernel(const float* __restrict__ in, _Float16* __restrict__ out, long n) {
  long i = (long)blockIdx.x * blockDim.x + threadIdx.x;
  if (i < n) out[i] = (_Float16)in[i];
}
__global__ void vec_add_kernel(const float* a, const float* b, float* o, int n) {
  int i = blockIdx.x * blockDim.x + threadIdx.x;
  if (i < n) o[i] = a[i] + b[i];
}
// convW (200,400) -> zero-padded f16 (224,416)
__global__ void convpad_kernel(const float* __restrict__ convW, _Float16* __restrict__ conv16) {
  int idx = blockIdx.x * blockDim.x + threadIdx.x;
  if (idx >= 224 * 416) return;
  int o = idx / 416, j = idx % 416;
  conv16[idx] = (o < 200 && j < 400) ? (_Float16)convW[o * 400 + j] : (_Float16)0;
}

// kmat16 (unmasked) and x16 (masked) build: (B,L,256)
__global__ void gather_kmat_kernel(const int* __restrict__ qs, const int* __restrict__ cs,
                                   const int* __restrict__ mask,
                                   const float* __restrict__ qemb, const float* __restrict__ iemb,
                                   _Float16* __restrict__ kmat16, _Float16* __restrict__ x16) {
  long idx = (long)blockIdx.x * blockDim.x + threadIdx.x;
  if (idx >= (long)32 * 200 * 256) return;
  int j = (int)(idx & 255); long bl = idx >> 8;
  int m = mask[bl];
  float v;
  if (j < 128) v = qemb[(long)qs[bl] * 128 + j];
  else         v = iemb[((long)cs[bl] + 256 * m) * 128 + (j - 128)];
  kmat16[idx] = (_Float16)v;
  x16[idx] = (_Float16)(v * (float)m);
}

// =====================================================================
// Persistent single-WGP LSTM (one direction). g = [x_t, h] @ [Wih;Whh]^T + bias.
// LDS: g16 (32x1024 f16, 64KB) | h16 (32x256 f16, 16KB) | c (32x256 f32, 32KB)
// 64 wave-tasks of 16x32 output per step over 32 waves (2-wide N blocking).
// =====================================================================
__global__ void lstm_kernel(const _Float16* __restrict__ x16,
                            const _Float16* __restrict__ Wih,
                            const _Float16* __restrict__ Whh,
                            const float* __restrict__ bias,
                            float* __restrict__ hs, int reverse) {
  extern __shared__ __align__(16) char smem[];
  _Float16* g16 = (_Float16*)smem;                       // 32*1024
  _Float16* h16 = (_Float16*)(smem + 65536);             // 32*256
  float*    cst = (float*)(smem + 65536 + 16384);        // 32*256
  int tid = threadIdx.x, wave = tid >> 5, l = tid & 31;
  for (int i = tid; i < 32 * 256; i += 1024) { h16[i] = (_Float16)0; cst[i] = 0.f; }
  __syncthreads();
  for (int t = 0; t < 200; ++t) {
    int tt = reverse ? (199 - t) : t;
    const _Float16* xt = x16 + (long)tt * 256;           // per-b row stride 200*256
    for (int task = wave; task < 64; task += 32) {
      int mt = task >> 5, ntp = task & 31;
      const _Float16* wih0 = Wih + (long)(ntp * 32) * 256;
      const _Float16* wih1 = wih0 + 16 * 256;
      const _Float16* whh0 = Whh + (long)(ntp * 32) * 256;
      const _Float16* whh1 = whh0 + 16 * 256;
      v8f acc0 = zero8(), acc1 = zero8();
      for (int k0 = 0; k0 < 256; k0 += 32) {
        __builtin_prefetch(wih0 + k0 + 64, 0, 1);
        v16h a = load_tile(xt, (long)200 * 256, mt * 16, k0);
        v16h b0 = load_tile(wih0, 256, 0, k0);
        v16h b1 = load_tile(wih1, 256, 0, k0);
        acc0 = wmma32(a, b0, acc0);
        acc1 = wmma32(a, b1, acc1);
      }
      for (int k0 = 0; k0 < 256; k0 += 32) {
        v16h a = load_tile(h16, 256, mt * 16, k0);
        v16h b0 = load_tile(whh0, 256, 0, k0);
        v16h b1 = load_tile(whh1, 256, 0, k0);
        acc0 = wmma32(a, b0, acc0);
        acc1 = wmma32(a, b1, acc1);
      }
      int col0 = ntp * 32 + (l & 15), rbase = mt * 16 + (l >> 4) * 8;
      float bv0 = bias[col0], bv1 = bias[col0 + 16];
#pragma unroll
      for (int r = 0; r < 8; ++r) {
        g16[(rbase + r) * 1024 + col0] = (_Float16)(acc0[r] + bv0);
        g16[(rbase + r) * 1024 + col0 + 16] = (_Float16)(acc1[r] + bv1);
      }
    }
    __syncthreads();
    for (int i = tid; i < 32 * 256; i += 1024) {
      int b = i >> 8, j = i & 255;
      float gi = (float)g16[b * 1024 + j];
      float gf = (float)g16[b * 1024 + 256 + j];
      float gg = (float)g16[b * 1024 + 512 + j];
      float go = (float)g16[b * 1024 + 768 + j];
      float c = sigf(gf) * cst[i] + sigf(gi) * tanhf(gg);
      float h = sigf(go) * tanhf(c);
      cst[i] = c; h16[i] = (_Float16)h;
      hs[((long)b * 200 + tt) * 256 + j] = h;
    }
    __syncthreads();
  }
}

__global__ void enc_build_kernel(const float* hf, const float* hb, _Float16* enc16, long n) {
  long i = (long)blockIdx.x * blockDim.x + threadIdx.x;
  if (i < n) enc16[i] = (_Float16)(hf[i] + hb[i]);
}

// q_ = stu @ Wa1^T + ba1  (32x128, tiny)
__global__ void qproj_kernel(const int* uid, const float* uemb, const float* Wa1,
                             const float* ba1, float* qf) {
  int i = blockIdx.x * blockDim.x + threadIdx.x;
  if (i >= 32 * 128) return;
  int b = i >> 7, j = i & 127;
  const float* s = uemb + (long)uid[b] * 128;
  const float* w = Wa1 + j * 128;
  float acc = ba1[j];
  for (int k = 0; k < 128; ++k) acc += s[k] * w[k];
  qf[i] = acc;
}

// Gumbel-softmax hard selection -> stu_deno in {0,1} (or -inf when masked out)
__global__ void alpha_kernel(const float* qf, const float* k1, const float* k2,
                             const float* Wa3, const float* ba3, const int* mask,
                             float* stu_deno) {
  int idx = blockIdx.x * blockDim.x + threadIdx.x;
  if (idx >= 6400) return;
  int b = idx / 200;
  const float* q = qf + b * 128;
  const float* r1 = k1 + (long)idx * 128;
  const float* r2 = k2 + (long)idx * 128;
  float a0 = ba3[0], a1 = ba3[1];
  for (int j = 0; j < 128; ++j) {
    float t1 = tanhf(q[j] + r1[j]);
    a0 += t1 * Wa3[j];       a1 += t1 * Wa3[256 + j];
  }
  for (int j = 0; j < 128; ++j) {
    float t2 = tanhf(q[j] + r2[j]);
    a0 += t2 * Wa3[128 + j]; a1 += t2 * Wa3[384 + j];
  }
  a0 = sigf(a0); a1 = sigf(a1);
  float g0 = -logf(-logf(u01(2u * idx + 1u) + 1e-10f) + 1e-10f);
  float g1 = -logf(-logf(u01(2u * idx + 2u) + 1e-10f) + 1e-10f);
  float hard = ((a1 + g1) > (a0 + g0)) ? 1.f : 0.f;   // y = yhard in fwd value
  stu_deno[idx] = mask[idx] ? hard : -INFINITY;
}

__global__ void zsample_kernel(const float* mu, const float* lv, _Float16* z16, long n) {
  long i = (long)blockIdx.x * blockDim.x + threadIdx.x;
  if (i < n) z16[i] = (_Float16)(mu[i] + __expf(lv[i] * 0.5f) * nrm((unsigned)i + 12345u));
}

// stFT (B, 256, 416): element (b, h, j) ; j=2c+w ; w=0 -> x, w=1 -> xr ; j>=400 -> 0
__global__ void stf_build_kernel(const _Float16* x16, const _Float16* xr16, _Float16* stFT) {
  long idx = (long)blockIdx.x * blockDim.x + threadIdx.x;
  if (idx >= (long)32 * 256 * 416) return;
  int j = (int)(idx % 416);
  long rem = idx / 416;
  int h = (int)(rem & 255);
  int b = (int)(rem >> 8);
  _Float16 v = (_Float16)0;
  if (j < 400) {
    int c = j >> 1, w = j & 1;
    long s = ((long)b * 200 + c) * 256 + h;
    v = w ? xr16[s] : x16[s];
  }
  stFT[idx] = v;
}

// score[b,o] = sigmoid( sum_h coT[b,h,o] * Wmlp[1,h] )
__global__ void score_kernel(const float* coT, const float* Wmlp, float* score) {
  int idx = blockIdx.x * blockDim.x + threadIdx.x;
  if (idx >= 6400) return;
  int b = idx / 200, o = idx % 200;
  const float* base = coT + (long)b * 256 * 224 + o;
  float acc = 0.f;
  for (int j = 0; j < 256; ++j) acc += base[(long)j * 224] * Wmlp[256 + j];
  score[idx] = sigf(acc);
}

// deno, ex/at/it embeddings, cat3 = [ex, at, a] (f16)
__global__ void deno_kernel(const float* score, const float* stu_deno,
                            const int* qs, const int* ut, const int* it, const int* corr,
                            const float* qemb, const float* uemb, const float* iemb,
                            float* exf, _Float16* it16, _Float16* cat3) {
  int idx = blockIdx.x * blockDim.x + threadIdx.x;
  if (idx >= 6400) return;
  float d = 1.f - score[idx] * stu_deno[idx];
  if (isinf(d) && d < 0.f) d = 0.f;                 // isneginf -> 0
  const float* eq = qemb + (long)qs[idx] * 128;
  const float* eu = uemb + (long)ut[idx] * 128;
  const float* ei = iemb + (long)it[idx] * 128;
  float a = (float)corr[idx];
  long o = (long)idx * 128;
  long o3 = (long)idx * 384;
  for (int k = 0; k < 128; ++k) {
    float ex = eq[k] * d, at = eu[k] * d, iv = ei[k] * d;
    exf[o + k] = ex;
    it16[o + k] = (_Float16)iv;
    cat3[o3 + k] = (_Float16)ex;
    cat3[o3 + 128 + k] = (_Float16)at;
    cat3[o3 + 256 + k] = (_Float16)a;
  }
}

__global__ void init_h_kernel(float* h, const float* h0) {
  long i = (long)blockIdx.x * blockDim.x + threadIdx.x;
  if (i < (long)32 * 256 * 128) h[i] = h0[i & 32767];
}
__global__ void htilde0_kernel(const float* qmat, const int* qs, const float* h0, float* htilde) {
  int i = blockIdx.x * blockDim.x + threadIdx.x;
  if (i >= 4096) return;
  int b = i >> 7, k = i & 127;
  const float* qr = qmat + (long)qs[b * 200] * 256;
  float acc = 0.f;
  for (int n = 0; n < 256; ++n) acc += qr[n] * h0[n * 128 + k];
  htilde[i] = acc;
}

// ---- scan step, part 1 (single WGP): IG/Gl/LG + gate_in ----
__global__ void scan_s1_kernel(const _Float16* allL16, const _Float16* it16, const float* htilde,
                               const _Float16* W2h, const _Float16* W3h,
                               const float* b2, const float* b3,
                               const _Float16* W4h, const float* b4,
                               float* LGf, float* gatef, int t) {
  __shared__ __align__(16) _Float16 x4[32 * 512];
  __shared__ __align__(16) _Float16 LG16[32 * 128];
  int tid = threadIdx.x, wave = tid >> 5, l = tid & 31;
  for (int i = tid; i < 32 * 512; i += 512) {
    int b = i >> 9, j = i & 511; float v;
    if (j < 128)      v = (t > 0) ? (float)allL16[((long)b * 200 + t - 1) * 128 + j] : 0.f;
    else if (j < 256) v = (float)it16[((long)b * 200 + t) * 128 + (j - 128)];
    else if (j < 384) v = (float)allL16[((long)b * 200 + t) * 128 + (j - 256)];
    else              v = htilde[b * 128 + (j - 384)];
    x4[i] = (_Float16)v;
  }
  __syncthreads();
  int mt = wave >> 3, nt = wave & 7;
  int col = nt * 16 + (l & 15), rbase = mt * 16 + (l >> 4) * 8;
  {
    v8f a2 = zero8(), a3 = zero8();
    for (int k0 = 0; k0 < 512; k0 += 32) {
      v16h a = load_tile(x4, 512, mt * 16, k0);
      v16h w2 = load_tile(W2h, 512, nt * 16, k0);
      v16h w3 = load_tile(W3h, 512, nt * 16, k0);
      a2 = wmma32(a, w2, a2);
      a3 = wmma32(a, w3, a3);
    }
#pragma unroll
    for (int r = 0; r < 8; ++r) {
      int row = rbase + r;
      float ig = tanhf(a2[r] + b2[col]);
      float gl = sigf(a3[r] + b3[col]);
      float lg = gl * (ig + 1.f) * 0.5f;
      LG16[row * 128 + col] = (_Float16)lg;
      LGf[row * 128 + col] = lg;
    }
  }
  __syncthreads();
  {
    v8f acc = zero8();
    for (int k0 = 0; k0 < 128; k0 += 32) {                 // LG @ W4l^T
      v16h a = load_tile(LG16, 128, mt * 16, k0);
      v16h w = load_tile(W4h + 128, 384, nt * 16, k0);
      acc = wmma32(a, w, acc);
    }
    const _Float16* itb = it16 + (long)t * 128;
    for (int k0 = 0; k0 < 128; k0 += 32) {                 // it @ W4i^T
      v16h a = load_tile(itb, (long)200 * 128, mt * 16, k0);
      v16h w = load_tile(W4h + 128, 384, nt * 16, 128 + k0);
      acc = wmma32(a, w, acc);
    }
#pragma unroll
    for (int r = 0; r < 8; ++r) gatef[(rbase + r) * 128 + col] = acc[r] + b4[col];
  }
}

// ---- scan step, part 2: gamma_f GEMM + h update + partial h_tilde ----
// 16 waves, each one 16x32 strip of the 64x128 tile (2-wide N blocking).
__global__ void scan_s2_kernel(float* hbuf, const _Float16* W4h16, const float* gatef,
                               const float* LGf, const float* qmat, const int* qseq,
                               float* partial, int t) {
  __shared__ __align__(16) _Float16 h16[64 * 128];
  __shared__ float pt[128];
  int b = blockIdx.x >> 2, nblk = blockIdx.x & 3;
  int tid = threadIdx.x, wave = tid >> 5, l = tid & 31;
  float* hp = hbuf + ((long)b * 256 + nblk * 64) * 128;
  for (int i = tid; i < 64 * 128; i += 512) h16[i] = (_Float16)hp[i];
  if (tid < 128) pt[tid] = 0.f;
  __syncthreads();
  const float* qr0 = qmat + (long)qseq[b * 200 + t] * 256;
  const float* qr1 = qmat + (long)qseq[b * 200 + t + 1] * 256;
  int mt = wave & 3, ntp = wave >> 2;                 // mt 0..3, ntp 0..3
  const _Float16* W0 = W4h16 + (long)(ntp * 32) * 384;
  const _Float16* W1 = W0 + 16 * 384;
  v8f acc0 = zero8(), acc1 = zero8();
  for (int k0 = 0; k0 < 128; k0 += 32) {
    v16h a = load_tile(h16, 128, mt * 16, k0);
    v16h w0 = load_tile(W0, 384, 0, k0);
    v16h w1 = load_tile(W1, 384, 0, k0);
    acc0 = wmma32(a, w0, acc0);
    acc1 = wmma32(a, w1, acc1);
  }
  int col0 = ntp * 32 + (l & 15), rbase = mt * 16 + (l >> 4) * 8;
  float gi0 = gatef[b * 128 + col0], gi1 = gatef[b * 128 + col0 + 16];
  float lg0 = LGf[b * 128 + col0],   lg1 = LGf[b * 128 + col0 + 16];
#pragma unroll
  for (int r = 0; r < 8; ++r) {
    int nl = rbase + r;
    int n = nblk * 64 + nl;
    float qe0 = qr0[n], qe1 = qr1[n];
    float hpv0 = hp[nl * 128 + col0];
    float hpv1 = hp[nl * 128 + col0 + 16];
    float hn0 = qe0 * lg0 + sigf(acc0[r] + gi0) * hpv0;
    float hn1 = qe0 * lg1 + sigf(acc1[r] + gi1) * hpv1;
    hp[nl * 128 + col0] = hn0;
    hp[nl * 128 + col0 + 16] = hn1;
    atomicAdd(&pt[col0], qe1 * hn0);          // ds_add_f32
    atomicAdd(&pt[col0 + 16], qe1 * hn1);
  }
  __syncthreads();
  if (tid < 128) partial[((long)b * 4 + nblk) * 128 + tid] = pt[tid];
}

// ---- scan step, part 3: h_tilde reduce + y output ----
__global__ void scan_s3_kernel(const float* partial, const float* exf, const float* W5,
                               const float* b5, float* htilde, float* out, int t) {
  __shared__ float sht[32 * 128];
  __shared__ float ysum[32];
  int tid = threadIdx.x;
  if (tid < 32) ysum[tid] = 0.f;
  for (int i = tid; i < 4096; i += 256) {
    float v = 0.f;
    for (int j = 0; j < 4; ++j) v += partial[(((long)(i >> 7)) * 4 + j) * 128 + (i & 127)];
    sht[i] = v; htilde[i] = v;
  }
  __syncthreads();
  for (int i = tid; i < 4096; i += 256) {
    int b = i >> 7, k = i & 127;
    float acc = b5[k];
    const float* ex = exf + ((long)b * 200 + t + 1) * 128;
    const float* w = W5 + k * 256;
    for (int j = 0; j < 128; ++j) acc += ex[j] * w[j];
    for (int j = 0; j < 128; ++j) acc += sht[b * 128 + j] * w[128 + j];
    atomicAdd(&ysum[b], sigf(acc));
  }
  __syncthreads();
  if (tid < 32) out[(long)tid * 199 + t] = ysum[tid] * (1.f / 128.f);
}

// =====================================================================
extern "C" void kernel_launch(void* const* d_in, const int* in_sizes, int n_in,
                              void* d_out, int out_size, void* d_ws, size_t ws_size,
                              hipStream_t stream) {
  (void)in_sizes; (void)n_in; (void)out_size; (void)ws_size;
  const int*   user_id = (const int*)d_in[0];
  const int*   qseq    = (const int*)d_in[1];
  const int*   cseq    = (const int*)d_in[2];
  const int*   corr    = (const int*)d_in[3];
  const int*   utseq   = (const int*)d_in[4];
  const int*   itseq   = (const int*)d_in[5];
  const int*   mask    = (const int*)d_in[6];
  const float* qmat    = (const float*)d_in[7];
  const float* h0      = (const float*)d_in[8];
  const float* user_emb   = (const float*)d_in[9];
  const float* quest_emb  = (const float*)d_in[10];
  const float* inter_emb  = (const float*)d_in[11];
  const float* ut_emb     = (const float*)d_in[12];
  const float* it_emb     = (const float*)d_in[13];
  const float* W1  = (const float*)d_in[14];
  const float* W2  = (const float*)d_in[15];
  const float* W3  = (const float*)d_in[16];
  const float* W4  = (const float*)d_in[17];
  const float* W5  = (const float*)d_in[18];
  const float* Wa1 = (const float*)d_in[19];
  const float* Wa2 = (const float*)d_in[20];
  const float* Wa3 = (const float*)d_in[21];
  const float* Wmlp = (const float*)d_in[22];
  const float* Wih_f = (const float*)d_in[23];
  const float* Whh_f = (const float*)d_in[24];
  const float* Wih_b = (const float*)d_in[25];
  const float* Whh_b = (const float*)d_in[26];
  const float* vW1 = (const float*)d_in[27];
  const float* vW2 = (const float*)d_in[28];
  const float* vW3 = (const float*)d_in[29];
  const float* vW4 = (const float*)d_in[30];
  const float* vW5 = (const float*)d_in[31];
  const float* b1 = (const float*)d_in[32];
  const float* b2 = (const float*)d_in[33];
  const float* b3 = (const float*)d_in[34];
  const float* b4 = (const float*)d_in[35];
  const float* b5 = (const float*)d_in[36];
  const float* ba1 = (const float*)d_in[37];
  const float* ba2 = (const float*)d_in[38];
  const float* ba3 = (const float*)d_in[39];
  const float* bih_f = (const float*)d_in[40];
  const float* bhh_f = (const float*)d_in[41];
  const float* bih_b = (const float*)d_in[42];
  const float* bhh_b = (const float*)d_in[43];
  const float* vb1 = (const float*)d_in[44];
  const float* vb2 = (const float*)d_in[45];
  const float* vb3 = (const float*)d_in[46];
  const float* vb4 = (const float*)d_in[47];
  const float* vb5 = (const float*)d_in[48];
  const float* convb = (const float*)d_in[49];
  const float* convW = (const float*)d_in[50];
  float* out = (float*)d_out;

  const long BL = 6400;          // 32 * 200
  char* base = (char*)d_ws;
  size_t off = 0;
  auto alloc = [&](size_t bytes) -> void* {
    void* p = base + off;
    off = (off + bytes + 255) & ~(size_t)255;
    return p;
  };
  // f16 weights
  _Float16* Wih_f16 = (_Float16*)alloc(1024 * 256 * 2);
  _Float16* Whh_f16 = (_Float16*)alloc(1024 * 256 * 2);
  _Float16* Wih_b16 = (_Float16*)alloc(1024 * 256 * 2);
  _Float16* Whh_b16 = (_Float16*)alloc(1024 * 256 * 2);
  _Float16* Wa2_16 = (_Float16*)alloc(128 * 256 * 2);
  _Float16* vW1_16 = (_Float16*)alloc(128 * 256 * 2);
  _Float16* vW2_16 = (_Float16*)alloc(128 * 128 * 2);
  _Float16* vW3_16 = (_Float16*)alloc(128 * 128 * 2);
  _Float16* vW4_16 = (_Float16*)alloc(128 * 128 * 2);
  _Float16* vW5_16 = (_Float16*)alloc(256 * 128 * 2);
  _Float16* W1_16 = (_Float16*)alloc(128 * 384 * 2);
  _Float16* W2_16 = (_Float16*)alloc(128 * 512 * 2);
  _Float16* W3_16 = (_Float16*)alloc(128 * 512 * 2);
  _Float16* W4_16 = (_Float16*)alloc(128 * 384 * 2);
  _Float16* conv16 = (_Float16*)alloc(224 * 416 * 2);
  float* lbias_f = (float*)alloc(1024 * 4);
  float* lbias_b = (float*)alloc(1024 * 4);
  // activations
  _Float16* kmat16 = (_Float16*)alloc(BL * 256 * 2);
  _Float16* x16 = (_Float16*)alloc(BL * 256 * 2);
  float* hf = (float*)alloc(BL * 256 * 4);
  float* hb = (float*)alloc(BL * 256 * 4);
  _Float16* enc16 = (_Float16*)alloc(BL * 256 * 2);
  float* k1f = (float*)alloc(BL * 128 * 4);
  float* k2f = (float*)alloc(BL * 128 * 4);
  float* qf = (float*)alloc(32 * 128 * 4);
  float* stu_deno = (float*)alloc(BL * 4);
  _Float16* vh16 = (_Float16*)alloc(BL * 128 * 2);
  float* muf = (float*)alloc(BL * 128 * 4);
  float* lvf = (float*)alloc(BL * 128 * 4);
  _Float16* z16 = (_Float16*)alloc(BL * 128 * 2);
  _Float16* h2_16 = (_Float16*)alloc(BL * 128 * 2);
  _Float16* xr16 = (_Float16*)alloc(BL * 256 * 2);
  _Float16* stFT = (_Float16*)alloc((long)32 * 256 * 416 * 2);
  float* coT = (float*)alloc((long)32 * 256 * 224 * 4);
  float* score = (float*)alloc(BL * 4);
  float* exf = (float*)alloc(BL * 128 * 4);
  _Float16* it16 = (_Float16*)alloc(BL * 128 * 2);
  _Float16* cat3 = (_Float16*)alloc(BL * 384 * 2);
  _Float16* allL16 = (_Float16*)alloc(BL * 128 * 2);
  float* hbuf = (float*)alloc((long)32 * 256 * 128 * 4);
  float* htilde = (float*)alloc(32 * 128 * 4);
  float* LGf = (float*)alloc(32 * 128 * 4);
  float* gatef = (float*)alloc(32 * 128 * 4);
  float* partial = (float*)alloc(32 * 4 * 128 * 4);

  auto cvt = [&](const float* s, _Float16* d, long n) {
    f32_to_f16_kernel<<<(unsigned)((n + 255) / 256), 256, 0, stream>>>(s, d, n);
  };
  cvt(Wih_f, Wih_f16, 1024 * 256); cvt(Whh_f, Whh_f16, 1024 * 256);
  cvt(Wih_b, Wih_b16, 1024 * 256); cvt(Whh_b, Whh_b16, 1024 * 256);
  cvt(Wa2, Wa2_16, 128 * 256);
  cvt(vW1, vW1_16, 128 * 256); cvt(vW2, vW2_16, 128 * 128);
  cvt(vW3, vW3_16, 128 * 128); cvt(vW4, vW4_16, 128 * 128);
  cvt(vW5, vW5_16, 256 * 128);
  cvt(W1, W1_16, 128 * 384); cvt(W2, W2_16, 128 * 512);
  cvt(W3, W3_16, 128 * 512); cvt(W4, W4_16, 128 * 384);
  convpad_kernel<<<(224 * 416 + 255) / 256, 256, 0, stream>>>(convW, conv16);
  vec_add_kernel<<<4, 256, 0, stream>>>(bih_f, bhh_f, lbias_f, 1024);
  vec_add_kernel<<<4, 256, 0, stream>>>(bih_b, bhh_b, lbias_b, 1024);

  // embeddings -> kmat16 / x16
  gather_kmat_kernel<<<(unsigned)((BL * 256 + 255) / 256), 256, 0, stream>>>(
      qseq, cseq, mask, quest_emb, inter_emb, kmat16, x16);

  // BiLSTM (one WGP per direction, state in LDS)
  const size_t lstm_lds = 65536 + 16384 + 32768;
  lstm_kernel<<<1, 1024, lstm_lds, stream>>>(x16, Wih_f16, Whh_f16, lbias_f, hf, 0);
  lstm_kernel<<<1, 1024, lstm_lds, stream>>>(x16, Wih_b16, Whh_b16, lbias_b, hb, 1);
  enc_build_kernel<<<(unsigned)((BL * 256 + 255) / 256), 256, 0, stream>>>(hf, hb, enc16, BL * 256);

  auto gemm = [&](const _Float16* A, long a_rs, long a_bs,
                  const _Float16* Wp, long w_ns, long w_bs,
                  const float* bias,
                  float* C, long c_rs, long c_bs,
                  _Float16* C16, long c16_rs, long c16_bs,
                  int M, int N, int Kd, int act, int batches) {
    int tiles = (M >> 4) * (N >> 5);
    dim3 grid((tiles + 15) / 16, batches);
    gemm_wmma_kernel<<<grid, 512, 0, stream>>>(A, a_rs, a_bs, Wp, w_ns, w_bs,
                                               bias, C, c_rs, c_bs,
                                               C16, c16_rs, c16_bs, M, N, Kd, act);
  };

  // attention keys
  gemm(enc16, 256, 0, Wa2_16, 256, 0, ba2, k1f, 128, 0, nullptr, 0, 0,
       (int)BL, 128, 256, 0, 1);
  gemm(kmat16, 256, 0, Wa2_16, 256, 0, ba2, k2f, 128, 0, nullptr, 0, 0,
       (int)BL, 128, 256, 0, 1);
  qproj_kernel<<<16, 256, 0, stream>>>(user_id, user_emb, Wa1, ba1, qf);
  alpha_kernel<<<25, 256, 0, stream>>>(qf, k1f, k2f, Wa3, ba3, mask, stu_deno);

  // VAE
  gemm(x16, 256, 0, vW1_16, 256, 0, vb1, nullptr, 0, 0, vh16, 128, 0,
       (int)BL, 128, 256, 1, 1);
  gemm(vh16, 128, 0, vW2_16, 128, 0, vb2, muf, 128, 0, nullptr, 0, 0,
       (int)BL, 128, 128, 0, 1);
  gemm(vh16, 128, 0, vW3_16, 128, 0, vb3, lvf, 128, 0, nullptr, 0, 0,
       (int)BL, 128, 128, 0, 1);
  zsample_kernel<<<(unsigned)((BL * 128 + 255) / 256), 256, 0, stream>>>(muf, lvf, z16, BL * 128);
  gemm(z16, 128, 0, vW4_16, 128, 0, vb4, nullptr, 0, 0, h2_16, 128, 0,
       (int)BL, 128, 128, 1, 1);
  gemm(h2_16, 128, 0, vW5_16, 128, 0, vb5, nullptr, 0, 0, xr16, 256, 0,
       (int)BL, 256, 128, 2, 1);
  stf_build_kernel<<<(unsigned)(((long)32 * 256 * 416 + 255) / 256), 256, 0, stream>>>(x16, xr16, stFT);
  // conv einsum (transposed): coT[b,h,o] = sum_j stFT[b,h,j]*conv16[o,j] + convb[o]
  gemm(stFT, 416, (long)256 * 416, conv16, 416, 0, convb,
       coT, 224, (long)256 * 224, nullptr, 0, 0, 256, 224, 416, 0, 32);
  score_kernel<<<25, 256, 0, stream>>>(coT, Wmlp, score);

  // deno + learning embeddings
  deno_kernel<<<25, 256, 0, stream>>>(score, stu_deno, qseq, utseq, itseq, corr,
                                      quest_emb, ut_emb, it_emb, exf, it16, cat3);
  gemm(cat3, 384, 0, W1_16, 384, 0, b1, nullptr, 0, 0, allL16, 128, 0,
       (int)BL, 128, 384, 0, 1);

  // scan init
  init_h_kernel<<<(unsigned)(((long)32 * 256 * 128 + 255) / 256), 256, 0, stream>>>(hbuf, h0);
  htilde0_kernel<<<16, 256, 0, stream>>>(qmat, qseq, h0, htilde);

  // 199 sequential steps
  for (int t = 0; t < 199; ++t) {
    scan_s1_kernel<<<1, 512, 0, stream>>>(allL16, it16, htilde, W2_16, W3_16, b2, b3,
                                          W4_16, b4, LGf, gatef, t);
    scan_s2_kernel<<<128, 512, 0, stream>>>(hbuf, W4_16, gatef, LGf, qmat, qseq, partial, t);
    scan_s3_kernel<<<1, 256, 0, stream>>>(partial, exf, W5, b5, htilde, out, t);
  }
}
